// RNNModel_68315749810459
// MI455X (gfx1250) — matrix operand order repaired and checked by reading, not compiled, not run on hardware
//
#include <hip/hip_runtime.h>
#include <hip/hip_bf16.h>

// ---------------------------------------------------------------------------
// Model dims
// ---------------------------------------------------------------------------
#define S_    128
#define B_    32
#define H_    1024
#define NIN_  1024
#define NTOK_ 32000
#define AW_   3
#define SB_   (S_ * B_)     // 4096
#define G4H_  (4 * H_)      // 4096

typedef __bf16 bf16;
typedef __attribute__((ext_vector_type(16))) __bf16 v16bf;
typedef __attribute__((ext_vector_type(8)))  float  v8f;
typedef __attribute__((ext_vector_type(4)))  unsigned int u32x4;

struct FragBytes { u32x4 lo, hi; };
static_assert(sizeof(FragBytes) == sizeof(v16bf), "frag size");

// ---------------------------------------------------------------------------
// Helpers
// ---------------------------------------------------------------------------
__device__ __forceinline__ bf16 f2bf(float x) {
  union { float f; unsigned u; } v; v.f = x;
  unsigned r = v.u + 0x7FFFu + ((v.u >> 16) & 1u);
  unsigned short h = (unsigned short)(r >> 16);
  return __builtin_bit_cast(bf16, h);
}

// A-matrix 16x32 bf16 frag (MxK, row-major, lda elems).
// lanes 0-15: row M=lane, elems[0..15]=K 0..15
// lanes 16-31: row M=lane-16, elems[0..15]=K 16..31
__device__ __forceinline__ v16bf load_a_frag(const bf16* A, int lda, int m0,
                                             int k, int lane) {
  int row = lane & 15;
  int hi  = lane >> 4;
  const bf16* p = A + (size_t)(m0 + row) * lda + (k + hi * 16);
  FragBytes f;
  f.lo = *(const u32x4*)(p);
  f.hi = *(const u32x4*)(p + 8);
  return __builtin_bit_cast(v16bf, f);
}

// B-matrix 32x16 bf16 frag from W stored (N,K) row-major (B[k][n]=W[n][k]).
// lanes 0-15: col N=lane, K=0..15 ; lanes 16-31: col N=lane-16, K=16..31
__device__ __forceinline__ v16bf load_b_frag(const bf16* W, int ldb, int n0,
                                             int k, int lane) {
  int col = lane & 15;
  int hi  = lane >> 4;
  const bf16* p = W + (size_t)(n0 + col) * ldb + (k + hi * 16);
  FragBytes f;
  f.lo = *(const u32x4*)(p);
  f.hi = *(const u32x4*)(p + 8);
  return __builtin_bit_cast(v16bf, f);
}

// Same B frag but sourced from an LDS panel laid out as [col][32 k-values].
__device__ __forceinline__ v16bf lds_b_frag(const bf16* panel, int t, int lane) {
  int col = lane & 15;
  int hi  = lane >> 4;
  const bf16* p = panel + (t * 16 + col) * 32 + hi * 16;
  FragBytes f;
  f.lo = *(const u32x4*)(p);
  f.hi = *(const u32x4*)(p + 8);
  return __builtin_bit_cast(v16bf, f);
}

__device__ __forceinline__ float sigmoidf_(float x) {
  return 1.0f / (1.0f + __expf(-x));
}

// ---------------------------------------------------------------------------
// Conversion / gather kernels
// ---------------------------------------------------------------------------
__global__ void cvt_f32_to_bf16(const float* __restrict__ src,
                                bf16* __restrict__ dst, size_t n) {
  size_t i = (size_t)blockIdx.x * blockDim.x + threadIdx.x;
  if (i < n) dst[i] = f2bf(src[i]);
}

__global__ void copy_f32(const float* __restrict__ src, float* __restrict__ dst,
                         size_t n) {
  size_t i = (size_t)blockIdx.x * blockDim.x + threadIdx.x;
  if (i < n) dst[i] = src[i];
}

// x[s,b,:] = emb[tokens[s,b], :]  (converted to bf16)
__global__ void embed_gather(const int* __restrict__ tokens,
                             const float* __restrict__ emb,
                             bf16* __restrict__ xbf, int nin) {
  size_t i = (size_t)blockIdx.x * blockDim.x + threadIdx.x;
  size_t total = (size_t)SB_ * nin;
  if (i >= total) return;
  size_t row = i / nin;
  size_t col = i - row * nin;
  int tok = tokens[row];
  xbf[i] = f2bf(emb[(size_t)tok * nin + col]);
}

// dst[c*R + r] = bf16(src[r*C + c])   (attn_W hk -> W'[n,k] layout)
__global__ void transpose_cvt(const float* __restrict__ src,
                              bf16* __restrict__ dst, int R, int C) {
  size_t i = (size_t)blockIdx.x * blockDim.x + threadIdx.x;
  if (i >= (size_t)R * C) return;
  int r = (int)(i / C);
  int c = (int)(i - (size_t)r * C);
  dst[(size_t)c * R + r] = f2bf(src[i]);
}

// ---------------------------------------------------------------------------
// Blocked GEMM with async-LDS B staging:
//   C[m,n] = sum_k A[m,k]*W[n,k] (+ bias1[n] + bias2[n])
// A: (M,K) bf16 row-major, W: (N,K) bf16 row-major, C: (M,N) fp32.
// Block = 8 waves = 128 M-rows sharing one 64-column B panel.
// B panel (64 cols x 32 k = 4KB) is double-buffered in LDS and filled with
// GLOBAL_LOAD_ASYNC_TO_LDS_B128 (one 512B async op per wave per K-chunk).
// Requires M % 128 == 0, N % 64 == 0, K % 32 == 0.
// ---------------------------------------------------------------------------
__global__ __launch_bounds__(256)
void gemm_bf16_stage(const bf16* __restrict__ A, const bf16* __restrict__ W,
                     const float* __restrict__ bias1, const float* __restrict__ bias2,
                     float* __restrict__ C, int M, int N, int K) {
  __shared__ __align__(16) bf16 bstage[2][64 * 32];   // [buf][col*32 + kk]

  int lane = threadIdx.x & 31;
  int wv   = threadIdx.x >> 5;           // 0..7
  int ngrp = N >> 6;                     // N / 64
  int mb   = blockIdx.x / ngrp;
  int ng   = blockIdx.x - mb * ngrp;
  int m0   = mb * 128 + wv * 16;
  int n0   = ng * 64;
  (void)M;

  // Staging map: wave wv moves cols [8*wv, 8*wv+8); lane l -> col 8*wv + (l>>2),
  // 16-byte k-chunk (l&3). Per async op: 32 lanes x 16B = 512B.
  int scol = wv * 8 + (lane >> 2);
  int skq  = (lane & 3) * 8;             // k offset in elements
  const bf16* gsrc = W + (size_t)(n0 + scol) * K + skq;
  unsigned lds_dst[2];
  lds_dst[0] = (unsigned)(uintptr_t)&bstage[0][scol * 32 + skq];
  lds_dst[1] = (unsigned)(uintptr_t)&bstage[1][scol * 32 + skq];

  v8f acc0 = {}, acc1 = {}, acc2 = {}, acc3 = {};
  int niter = K >> 5;

  // Prologue: stage chunk 0 into buffer 0.
  asm volatile("global_load_async_to_lds_b128 %0, %1, off"
               :: "v"(lds_dst[0]), "v"(gsrc) : "memory");

  for (int ki = 0; ki < niter; ++ki) {
    int cur = ki & 1;
    if (ki + 1 < niter) {
      const bf16* src = gsrc + ((ki + 1) << 5);
      asm volatile("global_load_async_to_lds_b128 %0, %1, off"
                   :: "v"(lds_dst[cur ^ 1]), "v"(src) : "memory");
      asm volatile("s_wait_asynccnt 0x1" ::: "memory");
    } else {
      asm volatile("s_wait_asynccnt 0x0" ::: "memory");
    }
    __syncthreads();   // all waves' writes for buffer `cur` have landed

    v16bf a = load_a_frag(A, K, m0, ki << 5, lane);
    const bf16* panel = &bstage[cur][0];
    v16bf b0 = lds_b_frag(panel, 0, lane);
    v16bf b1 = lds_b_frag(panel, 1, lane);
    v16bf b2 = lds_b_frag(panel, 2, lane);
    v16bf b3 = lds_b_frag(panel, 3, lane);
    acc0 = __builtin_amdgcn_wmma_f32_16x16x32_bf16(false, a, false, b0, (short)0, acc0, false, false);
    acc1 = __builtin_amdgcn_wmma_f32_16x16x32_bf16(false, a, false, b1, (short)0, acc1, false, false);
    acc2 = __builtin_amdgcn_wmma_f32_16x16x32_bf16(false, a, false, b2, (short)0, acc2, false, false);
    acc3 = __builtin_amdgcn_wmma_f32_16x16x32_bf16(false, a, false, b3, (short)0, acc3, false, false);

    __syncthreads();   // everyone done reading `cur` before it is overwritten
  }

  // C/D layout: lane<16 -> col=lane, rows m0+0..7 ; lane>=16 -> col=lane-16, rows m0+8..15
  int col = lane & 15;
  int hi  = lane >> 4;
  v8f accs[4] = {acc0, acc1, acc2, acc3};
#pragma unroll
  for (int t = 0; t < 4; ++t) {
    int n = n0 + t * 16 + col;
    float badd = 0.0f;
    if (bias1) badd += bias1[n];
    if (bias2) badd += bias2[n];
#pragma unroll
    for (int e = 0; e < 8; ++e) {
      int m = m0 + hi * 8 + e;
      __builtin_nontemporal_store(accs[t][e] + badd, &C[(size_t)m * N + n]);
    }
  }
}

// ---------------------------------------------------------------------------
// Fused LSTM step: gates = xg[s] + hprev @ Whh^T ; cell update.
// Grid: H/16 blocks x 256 threads (8 waves). Block handles 16 h-columns
// across all 4 gates and both 16-row halves of the batch (B=32).
// ---------------------------------------------------------------------------
__global__ __launch_bounds__(256)
void lstm_step(const float* __restrict__ xg,     // (B,4H) slice for step s
               const bf16* __restrict__ hprev,   // (B,H) bf16
               const bf16* __restrict__ whh,     // (4H,H) bf16
               float* __restrict__ cstate,       // (B,H) fp32, updated in place
               bf16*  __restrict__ ybf,          // (B,H) bf16 h output
               float* __restrict__ yf,           // optional (B,H) fp32 h output
               float* __restrict__ hN,           // optional final-h out
               float* __restrict__ cN) {         // optional final-c out
  __shared__ float g_lds[32][65];                // [b][gate*16 + hc]

  int lane = threadIdx.x & 31;
  int w    = threadIdx.x >> 5;   // 0..7
  int mt   = w & 1;
  int gate = w >> 1;             // 0..3
  int hc0  = blockIdx.x * 16;
  int n0   = gate * H_ + hc0;
  int m0   = mt * 16;

  v8f acc = {};
#pragma unroll 2
  for (int k = 0; k < H_; k += 32) {
    v16bf a = load_a_frag(hprev, H_, m0, k, lane);
    v16bf b = load_b_frag(whh, H_, n0, k, lane);
    acc = __builtin_amdgcn_wmma_f32_16x16x32_bf16(false, a, false, b, (short)0, acc, false, false);
  }

  int col = lane & 15;
  int hi  = lane >> 4;
#pragma unroll
  for (int e = 0; e < 8; ++e) {
    int b = m0 + hi * 8 + e;
    int n = n0 + col;
    g_lds[b][gate * 16 + col] = acc[e] + xg[(size_t)b * G4H_ + n];
  }
  __syncthreads();

  // Cell phase: 32 batches x 16 h-cols = 512 elements / 256 threads.
#pragma unroll
  for (int rep = 0; rep < 2; ++rep) {
    int p  = threadIdx.x + rep * 256;
    int b  = p >> 4;
    int hc = p & 15;
    int hg = hc0 + hc;
    float gi = sigmoidf_(g_lds[b][hc]);
    float gf = sigmoidf_(g_lds[b][16 + hc]);
    float gg = tanhf(g_lds[b][32 + hc]);
    float go = sigmoidf_(g_lds[b][48 + hc]);
    size_t idx = (size_t)b * H_ + hg;
    float c_new = gf * cstate[idx] + gi * gg;
    float h_new = go * tanhf(c_new);
    cstate[idx] = c_new;
    ybf[idx] = f2bf(h_new);
    if (yf) yf[idx] = h_new;
    if (hN) hN[idx] = h_new;
    if (cN) cN[idx] = c_new;
  }
}

// ---------------------------------------------------------------------------
// Attention pieces
// ---------------------------------------------------------------------------
// Z[m] = sum_k tanh(U[m,k]) * proj[k]   (one wave per row)
__global__ void attn_scores(const float* __restrict__ U,
                            const float* __restrict__ proj,
                            float* __restrict__ Z, int rows, int K) {
  int wid  = (int)(((size_t)blockIdx.x * blockDim.x + threadIdx.x) >> 5);
  int lane = threadIdx.x & 31;
  if (wid >= rows) return;
  const float* row = U + (size_t)wid * K;
  float sum = 0.0f;
  for (int k = lane; k < K; k += 32) sum += tanhf(row[k]) * proj[k];
#pragma unroll
  for (int off = 16; off > 0; off >>= 1) sum += __shfl_xor(sum, off, 32);
  if (lane == 0) Z[wid] = sum;
}

// Row softmax over the batch dim (reference softmaxes scores over axis=2 == b).
__global__ void softmax_over_b(const float* __restrict__ Z,
                               float* __restrict__ Zs, int rows) {
  int wid  = (int)(((size_t)blockIdx.x * blockDim.x + threadIdx.x) >> 5);
  int lane = threadIdx.x & 31;
  if (wid >= rows) return;
  float v = Z[(size_t)wid * B_ + lane];
  float m = v;
#pragma unroll
  for (int off = 16; off > 0; off >>= 1) m = fmaxf(m, __shfl_xor(m, off, 32));
  float e = __expf(v - m);
  float s = e;
#pragma unroll
  for (int off = 16; off > 0; off >>= 1) s += __shfl_xor(s, off, 32);
  Zs[(size_t)wid * B_ + lane] = e / s;
}

// out[s,b,h]: s<AW -> y1[s,b,h]; else ctx[w=s-AW] = sum_t Zs[w+t,b]*y1[w+t,b,h]
__global__ void attn_combine(const float* __restrict__ y1f,
                             const float* __restrict__ Zs,
                             bf16* __restrict__ outbf) {
  size_t i = (size_t)blockIdx.x * blockDim.x + threadIdx.x;
  size_t total = (size_t)SB_ * H_;
  if (i >= total) return;
  int h = (int)(i % H_);
  size_t sb = i / H_;
  int b = (int)(sb % B_);
  int s = (int)(sb / B_);
  float val;
  if (s < AW_) {
    val = y1f[i];
  } else {
    int w = s - AW_;
    val = 0.0f;
#pragma unroll
    for (int t = 0; t < AW_; ++t) {
      size_t r = (size_t)(w + t) * B_ + b;
      val += Zs[r] * y1f[r * H_ + h];
    }
  }
  outbf[i] = f2bf(val);
}

// ---------------------------------------------------------------------------
// Host: orchestration
// ---------------------------------------------------------------------------
static inline size_t align256(size_t x) { return (x + 255) & ~(size_t)255; }

extern "C" void kernel_launch(void* const* d_in, const int* in_sizes, int n_in,
                              void* d_out, int out_size, void* d_ws, size_t ws_size,
                              hipStream_t stream) {
  (void)in_sizes; (void)n_in; (void)out_size; (void)ws_size;
  const int*   tokens = (const int*)  d_in[0];
  const float* h0     = (const float*)d_in[1];
  const float* c0     = (const float*)d_in[2];
  const float* emb    = (const float*)d_in[3];
  const float* Wih0   = (const float*)d_in[4];
  const float* Whh0   = (const float*)d_in[5];
  const float* bih0   = (const float*)d_in[6];
  const float* bhh0   = (const float*)d_in[7];
  const float* Wih1   = (const float*)d_in[8];
  const float* Whh1   = (const float*)d_in[9];
  const float* bih1   = (const float*)d_in[10];
  const float* bhh1   = (const float*)d_in[11];
  const float* attnW  = (const float*)d_in[12];
  const float* attnP  = (const float*)d_in[13];
  const float* decW   = (const float*)d_in[14];
  const float* decB   = (const float*)d_in[15];
  float* out = (float*)d_out;

  // ---- workspace carve ----
  char* p = (char*)d_ws;
  auto carve = [&](size_t bytes) { char* r = p; p += align256(bytes); return r; };
  bf16*  xbf     = (bf16*) carve((size_t)SB_ * NIN_ * 2);      // gathered embeddings
  bf16*  wih0bf  = (bf16*) carve((size_t)G4H_ * NIN_ * 2);
  bf16*  whh0bf  = (bf16*) carve((size_t)G4H_ * H_ * 2);
  bf16*  wih1bf  = (bf16*) carve((size_t)G4H_ * H_ * 2);
  bf16*  whh1bf  = (bf16*) carve((size_t)G4H_ * H_ * 2);
  bf16*  decbf   = (bf16*) carve((size_t)NTOK_ * H_ * 2);
  bf16*  attnTbf = (bf16*) carve((size_t)H_ * H_ * 2);
  float* xg      = (float*)carve((size_t)SB_ * G4H_ * 4);      // reused L0 then L1
  bf16*  y0bf    = (bf16*) carve((size_t)SB_ * H_ * 2);
  bf16*  y1bf    = (bf16*) carve((size_t)SB_ * H_ * 2);
  float* y1f     = (float*)carve((size_t)SB_ * H_ * 4);
  float* U       = (float*)carve((size_t)SB_ * H_ * 4);
  float* Z       = (float*)carve((size_t)SB_ * 4);
  float* Zs      = (float*)carve((size_t)SB_ * 4);
  bf16*  outbf   = (bf16*) carve((size_t)SB_ * H_ * 2);
  bf16*  h0bf    = (bf16*) carve((size_t)2 * B_ * H_ * 2);
  float* cstate  = (float*)carve((size_t)2 * B_ * H_ * 4);

  auto cvt = [&](const float* s, bf16* d, size_t n) {
    cvt_f32_to_bf16<<<(unsigned)((n + 255) / 256), 256, 0, stream>>>(s, d, n);
  };
  auto gemm = [&](const bf16* A, const bf16* W, const float* b1, const float* b2,
                  float* C, int M, int N, int K) {
    int blocks = (M / 128) * (N / 64);
    gemm_bf16_stage<<<blocks, 256, 0, stream>>>(A, W, b1, b2, C, M, N, K);
  };

  // ---- 1. weight / input conversion ----
  cvt(Wih0, wih0bf, (size_t)G4H_ * NIN_);
  cvt(Whh0, whh0bf, (size_t)G4H_ * H_);
  cvt(Wih1, wih1bf, (size_t)G4H_ * H_);
  cvt(Whh1, whh1bf, (size_t)G4H_ * H_);
  cvt(decW, decbf,  (size_t)NTOK_ * H_);
  cvt(h0,   h0bf,   (size_t)2 * B_ * H_);
  copy_f32<<<(unsigned)((2 * B_ * H_ + 255) / 256), 256, 0, stream>>>(
      c0, cstate, (size_t)2 * B_ * H_);
  transpose_cvt<<<(unsigned)(((size_t)H_ * H_ + 255) / 256), 256, 0, stream>>>(
      attnW, attnTbf, H_, H_);
  embed_gather<<<(unsigned)(((size_t)SB_ * NIN_ + 255) / 256), 256, 0, stream>>>(
      tokens, emb, xbf, NIN_);

  const size_t DEC = (size_t)SB_ * NTOK_;

  // ---- 2. layer 0: input GEMM + scan ----
  gemm(xbf, wih0bf, bih0, bhh0, xg, SB_, G4H_, NIN_);
  for (int s = 0; s < S_; ++s) {
    const bf16* hprev = (s == 0) ? h0bf : y0bf + (size_t)(s - 1) * B_ * H_;
    float* hN = (s == S_ - 1) ? out + DEC + 0 * (size_t)B_ * H_ : nullptr;
    float* cN = (s == S_ - 1) ? out + DEC + 2 * (size_t)B_ * H_ : nullptr;
    lstm_step<<<H_ / 16, 256, 0, stream>>>(
        xg + (size_t)s * B_ * G4H_, hprev, whh0bf, cstate,
        y0bf + (size_t)s * B_ * H_, nullptr, hN, cN);
  }

  // ---- 3. layer 1: input GEMM + scan ----
  gemm(y0bf, wih1bf, bih1, bhh1, xg, SB_, G4H_, H_);
  for (int s = 0; s < S_; ++s) {
    const bf16* hprev = (s == 0) ? h0bf + (size_t)B_ * H_
                                 : y1bf + (size_t)(s - 1) * B_ * H_;
    float* hN = (s == S_ - 1) ? out + DEC + 1 * (size_t)B_ * H_ : nullptr;
    float* cN = (s == S_ - 1) ? out + DEC + 3 * (size_t)B_ * H_ : nullptr;
    lstm_step<<<H_ / 16, 256, 0, stream>>>(
        xg + (size_t)s * B_ * G4H_, hprev, whh1bf, cstate + (size_t)B_ * H_,
        y1bf + (size_t)s * B_ * H_, y1f + (size_t)s * B_ * H_, hN, cN);
  }

  // ---- 4. attention ----
  gemm(y1bf, attnTbf, nullptr, nullptr, U, SB_, H_, H_);          // U = y1 @ attn_W
  attn_scores<<<(SB_ * 32 + 255) / 256, 256, 0, stream>>>(U, attnP, Z, SB_, H_);
  softmax_over_b<<<(S_ * 32 + 255) / 256, 256, 0, stream>>>(Z, Zs, S_);
  attn_combine<<<(unsigned)(((size_t)SB_ * H_ + 255) / 256), 256, 0, stream>>>(
      y1f, Zs, outbf);

  // ---- 5. decoder (dominant GEMM, streams 524 MB out) ----
  gemm(outbf, decbf, decB, nullptr, out, SB_, NTOK_, H_);
}
